// Encoder_42786464203487
// MI455X (gfx1250) — matrix operand order repaired and checked by reading, not compiled, not run on hardware
//
#include <hip/hip_runtime.h>

#define N_NODES 20000
#define N_EDGES 100000
#define N_BATCH 256
#define DIM     32
#define HID     128
#define OUTD    1024   // DIM*DIM

typedef __attribute__((ext_vector_type(16))) _Float16 v16h;
typedef __attribute__((ext_vector_type(8)))  _Float16 v8h;
typedef __attribute__((ext_vector_type(8)))  float    v8f;

#define WMMA_F16(A, B, C) \
  __builtin_amdgcn_wmma_f32_16x16x32_f16(false, (A), false, (B), (short)0, (C), false, false)

__device__ __forceinline__ float sigf(float x) { return 1.0f / (1.0f + expf(-x)); }

// A-fragment (16x32 f16, M x K) from LDS, row-major rows of length ldK.
// CDNA5 layout: lanes 0-15 row M=lane hold K=kb+0..7 and kb+16..23;
//               lanes 16-31 row M=lane-16 hold K=kb+8..15 and kb+24..31.
__device__ __forceinline__ v16h load_a_frag_lds(const _Float16* base, int ldK, int kb) {
  int l   = threadIdx.x & 31;
  int row = l & 15;
  int k0  = (l < 16) ? kb : (kb + 8);
  const _Float16* p = base + row * ldK + k0;
  v8h lo = *reinterpret_cast<const v8h*>(p);
  v8h hi = *reinterpret_cast<const v8h*>(p + 16);
  v16h a;
#pragma unroll
  for (int j = 0; j < 8; ++j) { a[j] = lo[j]; a[8 + j] = hi[j]; }
  return a;
}

// B-fragment (32x16 f16, K x N) where weight w is row-major [col][k] (so B[k][n]=w[n][k]).
// Lanes 0-15: col = colBase+lane, K=kb+0..7 and kb+16..23;  lanes 16-31: K=kb+8..15 and kb+24..31.
__device__ __forceinline__ v16h load_b_frag(const _Float16* w, int K, int colBase, int kb) {
  int l   = threadIdx.x & 31;
  int col = colBase + (l & 15);
  int k0  = (l < 16) ? kb : (kb + 8);
  const _Float16* p = w + col * K + k0;
  v8h lo = *reinterpret_cast<const v8h*>(p);
  v8h hi = *reinterpret_cast<const v8h*>(p + 16);
  v16h b;
#pragma unroll
  for (int j = 0; j < 8; ++j) { b[j] = lo[j]; b[8 + j] = hi[j]; }
  return b;
}

// ---------------- utility kernels ----------------
__global__ void k_cvt_f16(const float* s, _Float16* d, int n) {
  int i = blockIdx.x * 256 + threadIdx.x;
  if (i < n) d[i] = (_Float16)s[i];
}
__global__ void k_zero_f32(float* p, int n) {
  int i = blockIdx.x * 256 + threadIdx.x;
  if (i < n) p[i] = 0.0f;
}

// out = relu(x @ lin0_w^T + b); h = out
__global__ void k_lin0(const float* x, const float* w, const float* b, float* out, float* h) {
  int i = blockIdx.x * 256 + threadIdx.x;
  if (i >= N_NODES * DIM) return;
  int n = i >> 5, j = i & 31;
  float acc = b[j];
  const float* xr = x + n * DIM;
  const float* wr = w + j * DIM;
#pragma unroll 8
  for (int k = 0; k < DIM; ++k) acc = fmaf(xr[k], wr[k], acc);
  float v = fmaxf(acc, 0.0f);
  out[i] = v;
  h[i]   = v;
}

// Edge network: hidden = relu(edge_attr @ w1^T + b1)  [16 x 128] per wave-tile,
// then ew = hidden @ w2^T + b2  [16 x 1024] via WMMA f16->f32.
// w2h is staged block-wide through LDS in 32KB chunks (8x reuse across waves).
// Output stored in WMMA-fragment-native layout:
//   ewf[ ((tile*64 + nt)*32 + lane)*8 + r ]  (one b128 store per lane per nt-tile)
__global__ void __launch_bounds__(256) k_edge(const float* eattr, const float* w1,
                                              const float* b1, const _Float16* w2h,
                                              const float* b2, _Float16* ewf) {
  __shared__ __align__(32) _Float16 hid_s[8][16 * HID];   // 32 KB
  __shared__ __align__(32) _Float16 w2s[128 * HID];       // 32 KB chunk: 128 cols x 128 K
  __shared__ float attr_s[8][16 * 5];
  int wave = threadIdx.x >> 5, lane = threadIdx.x & 31, tid = threadIdx.x;
  int tile = blockIdx.x * 8 + wave;
  bool active = (tile * 16 < N_EDGES);   // wave-uniform
  int e0 = tile * 16;

  v16h a0, a1, a2, a3;
  if (active) {
    for (int i = lane; i < 16 * 5; i += 32) attr_s[wave][i] = eattr[e0 * 5 + i];
    asm volatile("s_wait_dscnt 0" ::: "memory");
    for (int idx = lane; idx < 16 * HID; idx += 32) {
      int m = idx >> 7, k = idx & (HID - 1);
      float acc = b1[k];
#pragma unroll
      for (int j = 0; j < 5; ++j) acc = fmaf(attr_s[wave][m * 5 + j], w1[k * 5 + j], acc);
      hid_s[wave][m * HID + k] = (_Float16)fmaxf(acc, 0.0f);
    }
    asm volatile("s_wait_dscnt 0" ::: "memory");
    a0 = load_a_frag_lds(hid_s[wave], HID, 0);
    a1 = load_a_frag_lds(hid_s[wave], HID, 32);
    a2 = load_a_frag_lds(hid_s[wave], HID, 64);
    a3 = load_a_frag_lds(hid_s[wave], HID, 96);
  }

  int nl = lane & 15;
  for (int c = 0; c < 8; ++c) {
    __syncthreads();
    // cooperative block-wide copy of chunk c (cols c*128..c*128+127) into LDS
    const v8h* gsrc = reinterpret_cast<const v8h*>(w2h + (size_t)c * 128 * HID);
    v8h* ldst = reinterpret_cast<v8h*>(w2s);
#pragma unroll
    for (int idx = tid; idx < 2048; idx += 256) ldst[idx] = gsrc[idx];
    __syncthreads();
    if (active) {
      for (int q = 0; q < 8; ++q) {
        int nt = c * 8 + q;
        v8f cf = {};
        v16h bfr;
        bfr = load_b_frag(w2s, HID, q * 16, 0);  cf = WMMA_F16(a0, bfr, cf);
        bfr = load_b_frag(w2s, HID, q * 16, 32); cf = WMMA_F16(a1, bfr, cf);
        bfr = load_b_frag(w2s, HID, q * 16, 64); cf = WMMA_F16(a2, bfr, cf);
        bfr = load_b_frag(w2s, HID, q * 16, 96); cf = WMMA_F16(a3, bfr, cf);
        float bias = b2[nt * 16 + nl];
        v8h pk;
#pragma unroll
        for (int r = 0; r < 8; ++r) pk[r] = (_Float16)(cf[r] + bias);
        *reinterpret_cast<v8h*>(ewf + ((size_t)(tile * 64 + nt) * 32 + lane) * 8) = pk;
      }
    }
  }
}

__global__ void k_deg(const int* dst, float* deg) {
  int e = blockIdx.x * 256 + threadIdx.x;
  if (e < N_EDGES) atomicAdd(&deg[dst[e]], 1.0f);
}

// One wave per 16-edge tile, reading ew back in fragment layout:
// lane's fragment element r of nt-tile holds ew[edge=rowBase+r][col=nt*16+nl],
// where col = i*32 + o  ->  for nt=2i: o=nl, for nt=2i+1: o=nl+16.
// So each lane accumulates msg for 8 edges x 2 output dims; atomic scatter at end.
__global__ void __launch_bounds__(256) k_msg(const int* src, const int* dst,
                                             const float* out, const _Float16* ewf,
                                             float* agg) {
  __shared__ float xS[8][16 * DIM];
  __shared__ int sS[8][16];
  __shared__ int sD[8][16];
  int wave = threadIdx.x >> 5, lane = threadIdx.x & 31;
  int tile = blockIdx.x * 8 + wave;
  if (tile * 16 >= N_EDGES) return;
  int e0 = tile * 16;

  if (lane < 16) {
    sS[wave][lane] = src[e0 + lane];
    sD[wave][lane] = dst[e0 + lane];
  }
  asm volatile("s_wait_dscnt 0" ::: "memory");
  for (int idx = lane; idx < 16 * DIM; idx += 32) {
    int n = idx >> 5, j = idx & 31;
    xS[wave][idx] = out[sS[wave][n] * DIM + j];
  }
  asm volatile("s_wait_dscnt 0" ::: "memory");

  const v8h* fp = reinterpret_cast<const v8h*>(ewf + (size_t)tile * 64 * 256);
  // prefetch this tile's 32KB fragment block (1KB stride per lane)
  __builtin_prefetch(reinterpret_cast<const char*>(fp) + lane * 1024, 0, 1);

  int rowBase = (lane < 16) ? 0 : 8, nl = lane & 15;
  float acc0[8] = {0.f, 0.f, 0.f, 0.f, 0.f, 0.f, 0.f, 0.f};
  float acc1[8] = {0.f, 0.f, 0.f, 0.f, 0.f, 0.f, 0.f, 0.f};
#pragma unroll 4
  for (int i = 0; i < DIM; ++i) {
    v8h w0 = fp[(2 * i) * 32 + lane];
    v8h w1 = fp[(2 * i + 1) * 32 + lane];
#pragma unroll
    for (int r = 0; r < 8; ++r) {
      float xv = xS[wave][(rowBase + r) * DIM + i];
      acc0[r] = fmaf(xv, (float)w0[r], acc0[r]);
      acc1[r] = fmaf(xv, (float)w1[r], acc1[r]);
    }
  }
#pragma unroll
  for (int r = 0; r < 8; ++r) {
    int d = sD[wave][rowBase + r];
    atomicAdd(&agg[d * DIM + nl],      acc0[r]);
    atomicAdd(&agg[d * DIM + 16 + nl], acc1[r]);
  }
}

// One wave per 16-node tile: NNConv epilogue + feature_conv mix + GRU step, with
// gi/gh matvecs done as WMMA (16x32 A, 32x96 B split into 6 col tiles).
__global__ void __launch_bounds__(256) k_node(const float* agg, const float* deg,
                                              const float* conv_b, const float* fc_w,
                                              const float* fc_b, const _Float16* giw,
                                              const _Float16* ghw, const float* gib,
                                              const float* ghb, float* out, float* h,
                                              float* feat) {
  __shared__ __align__(32) _Float16 mS[8][16 * DIM];
  __shared__ __align__(32) _Float16 hS[8][16 * DIM];
  __shared__ float hF[8][16 * DIM];
  int wave = threadIdx.x >> 5, lane = threadIdx.x & 31;
  int tile = blockIdx.x * 8 + wave;
  if (tile * 16 >= N_NODES) return;
  int n0 = tile * 16;
  float fw0 = fc_w[0], fw1 = fc_w[1], fb = fc_b[0];

  for (int idx = lane; idx < 16 * DIM; idx += 32) {
    int n = n0 + (idx >> 5), j = idx & 31;
    float a  = agg[n * DIM + j] / fmaxf(deg[n], 1.0f);
    float mm = fmaxf(a + conv_b[j], 0.0f);
    float o  = out[n * DIM + j];
    float m2 = fw0 * o + fw1 * mm + fb;
    mS[wave][idx] = (_Float16)m2;
    float hv = h[n * DIM + j];
    hS[wave][idx] = (_Float16)hv;
    hF[wave][idx] = hv;
  }
  asm volatile("s_wait_dscnt 0" ::: "memory");

  v16h am = load_a_frag_lds(mS[wave], DIM, 0);
  v16h ah = load_a_frag_lds(hS[wave], DIM, 0);
  v8f gi[6], gh[6];
#pragma unroll
  for (int t = 0; t < 6; ++t) {
    v8f z = {};
    v16h b = load_b_frag(giw, DIM, t * 16, 0);
    gi[t] = WMMA_F16(am, b, z);
    b = load_b_frag(ghw, DIM, t * 16, 0);
    gh[t] = WMMA_F16(ah, b, z);
  }

  int nl = lane & 15, rowBase = (lane < 16) ? 0 : 8;
#pragma unroll
  for (int t = 0; t < 2; ++t) {
    float br_i = gib[t * 16 + nl],      br_h = ghb[t * 16 + nl];
    float bz_i = gib[32 + t * 16 + nl], bz_h = ghb[32 + t * 16 + nl];
    float bn_i = gib[64 + t * 16 + nl], bn_h = ghb[64 + t * 16 + nl];
#pragma unroll
    for (int r = 0; r < 8; ++r) {
      float rg = sigf(gi[t][r] + br_i + gh[t][r] + br_h);
      float zg = sigf(gi[t + 2][r] + bz_i + gh[t + 2][r] + bz_h);
      float ng = tanhf(gi[t + 4][r] + bn_i + rg * (gh[t + 4][r] + bn_h));
      int M = rowBase + r;
      int j = t * 16 + nl;
      float hv = hF[wave][M * DIM + j];
      float hn = (1.0f - zg) * ng + zg * hv;
      int gidx = (n0 + M) * DIM + j;
      out[gidx] = hn;
      h[gidx]   = hn;
      feat[gidx] = hn;
    }
  }
}

__global__ void k_layerconv(const float* f0, const float* f1, const float* f2,
                            const float* lc_w, const float* lc_b, float* gn) {
  int i = blockIdx.x * 256 + threadIdx.x;
  if (i < N_NODES * DIM)
    gn[i] = lc_w[0] * f0[i] + lc_w[1] * f1[i] + lc_w[2] * f2[i] + lc_b[0];
}

__global__ void k_counts(const int* batch, int* counts) {
  int n = blockIdx.x * 256 + threadIdx.x;
  if (n < N_NODES) atomicAdd(&counts[batch[n]], 1);
}
__global__ void k_scan(const int* counts, int* offsets) {
  if (threadIdx.x == 0 && blockIdx.x == 0) {
    int s = 0;
    for (int b = 0; b < N_BATCH; ++b) { offsets[b] = s; s += counts[b]; }
    offsets[N_BATCH] = s;
  }
}

// One wave per graph: Set2Set LSTM cell (gate order i,f,g,o).
__global__ void k_lstm(const float* wih, const float* whh, const float* bih,
                       const float* bhh, const float* q_star, float* hh, float* cc) {
  int b = blockIdx.x, lane = threadIdx.x;
  float g4[4];
#pragma unroll
  for (int q = 0; q < 4; ++q) {
    int j = q * DIM + lane;
    float acc = bih[j] + bhh[j];
    for (int k = 0; k < 2 * DIM; ++k) acc = fmaf(q_star[b * 2 * DIM + k], wih[j * 2 * DIM + k], acc);
    for (int k = 0; k < DIM; ++k)     acc = fmaf(hh[b * DIM + k], whh[j * DIM + k], acc);
    g4[q] = acc;
  }
  float ig = sigf(g4[0]), fg = sigf(g4[1]);
  float gg = tanhf(g4[2]), og = sigf(g4[3]);
  float c  = fg * cc[b * DIM + lane] + ig * gg;
  float hn = og * tanhf(c);
  cc[b * DIM + lane] = c;
  hh[b * DIM + lane] = hn;
}

// One block per graph: segment softmax attention + weighted readout r; q_star = [hh, r].
__global__ void __launch_bounds__(256) k_attn(const float* gn, const float* hh,
                                              const int* offsets, float* e_arr,
                                              float* q_star) {
  int b = blockIdx.x, t = threadIdx.x;
  __shared__ float red[256];
  __shared__ float r_s[DIM];
  __shared__ float hv[DIM];
  if (t < DIM) { hv[t] = hh[b * DIM + t]; r_s[t] = 0.0f; }
  __syncthreads();
  int n0 = offsets[b], n1 = offsets[b + 1];

  float lmax = -INFINITY;
  for (int n = n0 + t; n < n1; n += 256) {
    float acc = 0.0f;
#pragma unroll
    for (int j = 0; j < DIM; ++j) acc = fmaf(gn[n * DIM + j], hv[j], acc);
    e_arr[n] = acc;
    lmax = fmaxf(lmax, acc);
  }
  red[t] = lmax;
  __syncthreads();
  for (int s = 128; s > 0; s >>= 1) { if (t < s) red[t] = fmaxf(red[t], red[t + s]); __syncthreads(); }
  float emax = red[0];
  if (!(emax > -INFINITY)) emax = 0.0f;  // non-finite -> 0 (matches reference)
  __syncthreads();

  float lsum = 0.0f;
  for (int n = n0 + t; n < n1; n += 256) {
    float ex = expf(e_arr[n] - emax);
    e_arr[n] = ex;
    lsum += ex;
  }
  red[t] = lsum;
  __syncthreads();
  for (int s = 128; s > 0; s >>= 1) { if (t < s) red[t] += red[t + s]; __syncthreads(); }
  float denom = fmaxf(red[0], 1e-16f);
  __syncthreads();

  for (int n = n0 + t; n < n1; n += 256) {
    float a = e_arr[n] / denom;
#pragma unroll
    for (int j = 0; j < DIM; ++j) atomicAdd(&r_s[j], a * gn[n * DIM + j]);
  }
  __syncthreads();
  if (t < DIM) {
    q_star[b * 2 * DIM + t]       = hv[t];
    q_star[b * 2 * DIM + DIM + t] = r_s[t];
  }
}

extern "C" void kernel_launch(void* const* d_in, const int* in_sizes, int n_in,
                              void* d_out, int out_size, void* d_ws, size_t ws_size,
                              hipStream_t stream) {
  (void)in_sizes; (void)n_in; (void)out_size; (void)ws_size;
  const float* x        = (const float*)d_in[0];
  const int*   ei       = (const int*)d_in[1];
  const float* eattr    = (const float*)d_in[2];
  const int*   batch    = (const int*)d_in[3];
  const float* lin0_w   = (const float*)d_in[4];
  const float* lin0_b   = (const float*)d_in[5];
  const float* enn_w1   = (const float*)d_in[6];
  const float* enn_b1   = (const float*)d_in[7];
  const float* enn_w2   = (const float*)d_in[8];
  const float* enn_b2   = (const float*)d_in[9];
  const float* conv_b   = (const float*)d_in[10];
  const float* fc_w     = (const float*)d_in[11];
  const float* fc_b     = (const float*)d_in[12];
  const float* gru_w_ih = (const float*)d_in[13];
  const float* gru_w_hh = (const float*)d_in[14];
  const float* gru_b_ih = (const float*)d_in[15];
  const float* gru_b_hh = (const float*)d_in[16];
  const float* lc_w     = (const float*)d_in[17];
  const float* lc_b     = (const float*)d_in[18];
  const float* ls_w_ih  = (const float*)d_in[19];
  const float* ls_w_hh  = (const float*)d_in[20];
  const float* ls_b_ih  = (const float*)d_in[21];
  const float* ls_b_hh  = (const float*)d_in[22];
  const int* src = ei;
  const int* dst = ei + N_EDGES;

  char* base = (char*)d_ws;
  size_t off = 0;
  auto alloc = [&](size_t bytes) -> void* {
    void* p = base + off;
    off += (bytes + 255) & ~(size_t)255;
    return p;
  };
  _Float16* w2h    = (_Float16*)alloc((size_t)OUTD * HID * 2);
  _Float16* giwh   = (_Float16*)alloc((size_t)3 * DIM * DIM * 2);
  _Float16* ghwh   = (_Float16*)alloc((size_t)3 * DIM * DIM * 2);
  _Float16* ewf    = (_Float16*)alloc((size_t)N_EDGES * OUTD * 2);  // fragment layout
  float*    deg    = (float*)alloc((size_t)N_NODES * 4);
  float*    outbuf = (float*)alloc((size_t)N_NODES * DIM * 4);
  float*    hbuf   = (float*)alloc((size_t)N_NODES * DIM * 4);
  float*    aggbuf = (float*)alloc((size_t)N_NODES * DIM * 4);
  float*    feats  = (float*)alloc((size_t)3 * N_NODES * DIM * 4);
  float*    gnbuf  = (float*)alloc((size_t)N_NODES * DIM * 4);
  float*    hhbuf  = (float*)alloc((size_t)N_BATCH * DIM * 4);
  float*    ccbuf  = (float*)alloc((size_t)N_BATCH * DIM * 4);
  float*    e_arr  = (float*)alloc((size_t)N_NODES * 4);
  int*      counts = (int*)alloc((size_t)N_BATCH * 4);
  int*      offs   = (int*)alloc((size_t)(N_BATCH + 1) * 4);
  float*    q_star = (float*)d_out;  // [B, 2*DIM] final output

  const int ETILES = N_EDGES / 16;   // 6250
  const int NTILES = N_NODES / 16;   // 1250

  // init / weight conversion
  k_zero_f32<<<(N_NODES + 255) / 256, 256, 0, stream>>>(deg, N_NODES);
  k_zero_f32<<<1, 256, 0, stream>>>((float*)counts, N_BATCH);
  k_zero_f32<<<(N_BATCH * 2 * DIM + 255) / 256, 256, 0, stream>>>(q_star, N_BATCH * 2 * DIM);
  k_zero_f32<<<(N_BATCH * DIM + 255) / 256, 256, 0, stream>>>(hhbuf, N_BATCH * DIM);
  k_zero_f32<<<(N_BATCH * DIM + 255) / 256, 256, 0, stream>>>(ccbuf, N_BATCH * DIM);
  k_cvt_f16<<<(OUTD * HID + 255) / 256, 256, 0, stream>>>(enn_w2, w2h, OUTD * HID);
  k_cvt_f16<<<(3 * DIM * DIM + 255) / 256, 256, 0, stream>>>(gru_w_ih, giwh, 3 * DIM * DIM);
  k_cvt_f16<<<(3 * DIM * DIM + 255) / 256, 256, 0, stream>>>(gru_w_hh, ghwh, 3 * DIM * DIM);

  // lin0 + relu ; edge network (WMMA GEMM) ; degree ; CSR offsets
  k_lin0<<<(N_NODES * DIM + 255) / 256, 256, 0, stream>>>(x, lin0_w, lin0_b, outbuf, hbuf);
  k_edge<<<(ETILES + 7) / 8, 256, 0, stream>>>(eattr, enn_w1, enn_b1, w2h, enn_b2, ewf);
  k_deg<<<(N_EDGES + 255) / 256, 256, 0, stream>>>(dst, deg);
  k_counts<<<(N_NODES + 255) / 256, 256, 0, stream>>>(batch, counts);
  k_scan<<<1, 32, 0, stream>>>(counts, offs);

  // 3 message-passing + GRU layers
  for (int layer = 0; layer < 3; ++layer) {
    k_zero_f32<<<(N_NODES * DIM + 255) / 256, 256, 0, stream>>>(aggbuf, N_NODES * DIM);
    k_msg<<<(ETILES + 7) / 8, 256, 0, stream>>>(src, dst, outbuf, ewf, aggbuf);
    k_node<<<(NTILES + 7) / 8, 256, 0, stream>>>(
        aggbuf, deg, conv_b, fc_w, fc_b, giwh, ghwh, gru_b_ih, gru_b_hh,
        outbuf, hbuf, feats + (size_t)layer * N_NODES * DIM);
  }
  k_layerconv<<<(N_NODES * DIM + 255) / 256, 256, 0, stream>>>(
      feats, feats + (size_t)N_NODES * DIM, feats + (size_t)2 * N_NODES * DIM,
      lc_w, lc_b, gnbuf);

  // Set2Set: 3 processing steps
  for (int it = 0; it < 3; ++it) {
    k_lstm<<<N_BATCH, 32, 0, stream>>>(ls_w_ih, ls_w_hh, ls_b_ih, ls_b_hh,
                                       q_star, hhbuf, ccbuf);
    k_attn<<<N_BATCH, 256, 0, stream>>>(gnbuf, hhbuf, offs, e_arr, q_star);
  }
}
